// LP_Gabor_Layer_44822278701716
// MI455X (gfx1250) — matrix operand order repaired and checked by reading, not compiled, not run on hardware
//
#include <hip/hip_runtime.h>
#include <hip/hip_bf16.h>

typedef __attribute__((ext_vector_type(16))) _Float16 v16h;
typedef __attribute__((ext_vector_type(8)))  float    v8f;

#define B_   16
#define C_   3
#define H_   192
#define W_   192
#define OC_  128
#define KG_  11      // gabor kernel
#define KL_  5       // lowpass kernel
#define SPAD 16      // s (kx) padded 11 -> 16
#define NROW (C_ * KG_)        // 33 (c,ky) rows
#define KPAD (NROW * SPAD)     // 528 padded K
#define NCHUNK 17              // ceil(528/32)
#define NTILE  8               // 128/16 channel tiles
#define PX   32                // positions per block (2 N-tiles per wave)

// ---------------------------------------------------------------------------
// Kernel 1: pack Gabor weights (128,3,11,11) f32 -> f16 WMMA A-fragments.
// K' = (c*11 + ky)*16 + s  (s in 0..15; s>=11 or K'>=528 => weight 0).
// A-fragment (16-bit A 16x32, contiguous k = 16*hi + i): lanes 0-15 hold K{0..15},
// lanes 16-31 hold K{16..31}; row M = tile*16 + (lane&15).
// Buffer index = ((chunk*8 + tile)*32 + lane)*16 halves (32B/lane).
// ---------------------------------------------------------------------------
__global__ __launch_bounds__(256) void pack_gabor_kernel(
    const float* __restrict__ gw, _Float16* __restrict__ apack) {
  int chunk = blockIdx.x;                 // 0..16
  int rem   = threadIdx.x;                // 0..255
  int tile  = rem >> 5;
  int lane  = rem & 31;
  int M     = tile * 16 + (lane & 15);
  int hi    = lane >> 4;
  int kbase = chunk * 32 + hi * 16;
  _Float16* dst = apack + (((size_t)chunk * 256 + rem) << 4);
#pragma unroll
  for (int i = 0; i < 16; ++i) {
    int k = kbase + i;
    float v = 0.0f;
    if (k < KPAD) {
      int row = k >> 4;                   // (c*11 + ky)
      int s   = k & 15;
      if (s < KG_) {
        int c  = row / KG_;
        int ky = row - c * KG_;
        v = gw[((M * C_ + c) * KG_ + ky) * KG_ + s];
      }
    }
    dst[i] = (_Float16)v;
  }
}

// ---------------------------------------------------------------------------
// Kernel 2: depthwise 5x5 low-pass + noise + soft-threshold, store f16.
// thr = (8/255) * mean_over_outch(sum|lp_weight|).
// ---------------------------------------------------------------------------
__global__ __launch_bounds__(256) void lp_quant_kernel(
    const float* __restrict__ x, const float* __restrict__ noise,
    const float* __restrict__ lpw, _Float16* __restrict__ q) {
  __shared__ float s_thr;
  if (threadIdx.x == 0) {
    float s = 0.0f;
    for (int i = 0; i < C_ * KL_ * KL_; ++i) s += fabsf(lpw[i]);
    s_thr = (8.0f / 255.0f) * (s / (float)C_);
  }
  __syncthreads();
  int gid = blockIdx.x * 256 + threadIdx.x;          // N divisible by 256
  int xx = gid % W_;
  int t  = gid / W_;
  int yy = t % H_;  t /= H_;
  int c  = t % C_;
  int b  = t / C_;
  const float* xin = x + (b * C_ + c) * (H_ * W_);
  const float* wc  = lpw + c * (KL_ * KL_);
  float o = 0.0f;
#pragma unroll
  for (int ky = 0; ky < KL_; ++ky) {
    int iy = yy + ky - 2;
    if (iy < 0 || iy >= H_) continue;
#pragma unroll
    for (int kx = 0; kx < KL_; ++kx) {
      int ix = xx + kx - 2;
      if (ix < 0 || ix >= W_) continue;
      o += xin[iy * W_ + ix] * wc[ky * KL_ + kx];
    }
  }
  o += noise[gid] * (16.0f / 255.0f) - (8.0f / 255.0f);
  float a = fabsf(o) - s_thr;
  float r = (a > 0.0f) ? ((o > 0.0f) ? a : -a) : 0.0f;
  q[gid] = (_Float16)r;
}

// ---------------------------------------------------------------------------
// Kernel 3: implicit-GEMM 11x11 conv 3->128 via v_wmma_f32_16x16x32_f16,
// N-blocked x2 (one A fragment feeds two WMMAs), then per-position
// argmax-channel mask + soft-threshold.
// Block = 256 threads = 8 waves; wave w owns channels [16w, 16w+16).
// Block covers 32 consecutive x positions at one (b, y).
// ---------------------------------------------------------------------------
__global__ __launch_bounds__(256) void gabor_wmma_kernel(
    const _Float16* __restrict__ q, const _Float16* __restrict__ apack,
    const float* __restrict__ lpw, float* __restrict__ out) {
  // Compact zero-padded patch: rows (c,ky), cols x0-5 .. x0+42.
  __shared__ _Float16 patch[NROW][48];
  // Expanded im2col: [exrow = 2*chunk+hi][col 0..31][16 halves]; row 33 = 0.
  __shared__ __attribute__((aligned(32))) _Float16 expatch[NROW + 1][PX][16];
  __shared__ float red0[NTILE][32];     // |max| for cols 0..15
  __shared__ float red1[NTILE][32];     // |max| for cols 16..31
  __shared__ float posmax[PX];
  __shared__ float s_thr;

  const int tid  = threadIdx.x;
  const int lane = tid & 31;
  const int wave = tid >> 5;
  const int x0   = blockIdx.x * PX;     // 6 tiles
  const int y    = blockIdx.y;          // 192
  const int b    = blockIdx.z;          // 16

  if (tid == 0) {
    float s = 0.0f;
    for (int i = 0; i < C_ * KL_ * KL_; ++i) s += fabsf(lpw[i]);
    s_thr = (8.0f / 255.0f) * (s / (float)C_);
  }
  // Stage 1: compact patch (33*48 = 1584 halves), zero padded at borders.
  for (int e = tid; e < NROW * 48; e += 256) {
    int row = e / 48;                   // (c*11 + ky)
    int xi  = e - row * 48;
    int c   = row / KG_;
    int ky  = row - c * KG_;
    int yy  = y + ky - 5;
    int xx  = x0 + xi - 5;
    _Float16 v = (_Float16)0.0f;
    if (yy >= 0 && yy < H_ && xx >= 0 && xx < W_)
      v = q[((b * C_ + c) * H_ + yy) * W_ + xx];
    patch[row][xi] = v;
  }
  __syncthreads();
  // Stage 2: expand to im2col vectors (34*32*16 = 17408 halves).
  for (int e = tid; e < (NROW + 1) * PX * 16; e += 256) {
    int i     = e & 15;
    int vec   = e >> 4;
    int col   = vec & (PX - 1);
    int exrow = vec >> 5;
    _Float16 v = (_Float16)0.0f;
    if (exrow < NROW) v = patch[exrow][col + i];   // col+i <= 46 < 48
    expatch[exrow][col][i] = v;
  }
  __syncthreads();

  const int col = lane & 15;            // position within N-tile
  const int hi  = lane >> 4;
  const _Float16* abase = apack + ((size_t)(wave * 32 + lane) << 4);
  v8f acc0 = {};
  v8f acc1 = {};
#pragma unroll
  for (int chunk = 0; chunk < NCHUNK; ++chunk) {
    v16h afrag  = *(const v16h*)(abase + ((size_t)chunk << 12)); // chunk*256*16
    v16h bfrag0 = *(const v16h*)(&expatch[2 * chunk + hi][col][0]);
    v16h bfrag1 = *(const v16h*)(&expatch[2 * chunk + hi][col + 16][0]);
    acc0 = __builtin_amdgcn_wmma_f32_16x16x32_f16(
        false, afrag, false, bfrag0, (short)0, acc0, false, false);
    acc1 = __builtin_amdgcn_wmma_f32_16x16x32_f16(
        false, afrag, false, bfrag1, (short)0, acc1, false, false);
  }

  // take_top_coeff: per-position max|coeff| over all 128 channels.
  float amax0 = 0.0f, amax1 = 0.0f;
#pragma unroll
  for (int r = 0; r < 8; ++r) {
    amax0 = fmaxf(amax0, fabsf(acc0[r]));
    amax1 = fmaxf(amax1, fabsf(acc1[r]));
  }
  red0[wave][lane] = amax0;
  red1[wave][lane] = amax1;
  __syncthreads();
  if (tid < PX) {
    float m = 0.0f;
    if (tid < 16) {
#pragma unroll
      for (int w = 0; w < NTILE; ++w)
        m = fmaxf(m, fmaxf(red0[w][tid], red0[w][tid + 16]));
    } else {
#pragma unroll
      for (int w = 0; w < NTILE; ++w)
        m = fmaxf(m, fmaxf(red1[w][tid - 16], red1[w][tid]));
    }
    posmax[tid] = m;
  }
  __syncthreads();

  const float m0  = posmax[col];
  const float m1  = posmax[col + 16];
  const float thr = s_thr;
  // C/D layout: VGPR r, lanes 0-15 -> M=r, lanes 16-31 -> M=r+8.
#pragma unroll
  for (int r = 0; r < 8; ++r) {
    int ch = wave * 16 + hi * 8 + r;
    size_t base = ((size_t)(b * OC_ + ch) * H_ + y) * W_;
    {
      float v    = acc0[r];
      float kept = (fabsf(v) == m0) ? v : 0.0f;
      float a    = fabsf(kept) - thr;
      float o    = (a > 0.0f) ? ((kept > 0.0f) ? a : -a) : 0.0f;
      out[base + x0 + col] = o;
    }
    {
      float v    = acc1[r];
      float kept = (fabsf(v) == m1) ? v : 0.0f;
      float a    = fabsf(kept) - thr;
      float o    = (a > 0.0f) ? ((kept > 0.0f) ? a : -a) : 0.0f;
      out[base + x0 + col + 16] = o;
    }
  }
}

// ---------------------------------------------------------------------------
extern "C" void kernel_launch(void* const* d_in, const int* in_sizes, int n_in,
                              void* d_out, int out_size, void* d_ws, size_t ws_size,
                              hipStream_t stream) {
  const float* x     = (const float*)d_in[0];   // (16,3,192,192)
  const float* noise = (const float*)d_in[1];   // (16,3,192,192)
  const float* lpw   = (const float*)d_in[2];   // (3,1,5,5)
  const float* gw    = (const float*)d_in[3];   // (128,3,11,11)
  float* out = (float*)d_out;                   // (16,128,192,192)

  // Workspace: [A-pack f16: 17*8*32*16 halves = 136KB][q f16: 3.4MB]
  _Float16* apack = (_Float16*)d_ws;
  _Float16* q     = (_Float16*)((char*)d_ws + (size_t)(NCHUNK * NTILE * 32 * 16) * 2);

  pack_gabor_kernel<<<NCHUNK, 256, 0, stream>>>(gw, apack);

  int n = B_ * C_ * H_ * W_;                    // 1,769,472 (divisible by 256)
  lp_quant_kernel<<<n / 256, 256, 0, stream>>>(x, noise, lpw, q);

  dim3 grid(W_ / PX, H_, B_);                   // (6, 192, 16)
  gabor_wmma_kernel<<<grid, 256, 0, stream>>>(q, apack, lpw, out);
}
